// MLA_18004502905038
// MI455X (gfx1250) — matrix operand order repaired and checked by reading, not compiled, not run on hardware
//
#include <hip/hip_runtime.h>
#include <hip/hip_bf16.h>
#include <cstdint>

typedef __attribute__((ext_vector_type(16))) _Float16 v16h;
typedef __attribute__((ext_vector_type(8)))  float    v8f;
typedef __attribute__((ext_vector_type(4)))  uint32_t u32x4;
typedef __attribute__((ext_vector_type(8)))  int      i32x8;
typedef __attribute__((ext_vector_type(4)))  int      i32x4;

#define B_    2
#define S_    2048
#define DIM_  2048
#define H_    16
#define QLORA 1536
#define KVLORA 512
#define NOPE_ 128
#define QKH   192   // NOPE + ROPE

__device__ __forceinline__ _Float16 conv_h(float x)    { return (_Float16)x; }
__device__ __forceinline__ _Float16 conv_h(_Float16 x) { return x; }

// ---------------------------------------------------------------------------
// Generic WMMA GEMM: C[M,N] = A[M,K] * op(B), f32/f16 operands converted to f16
// in LDS.  TRANS_B: B = W[N,K] (C = A W^T).  !TRANS_B: B = Bm[K,N] (C = A Bm).
// Grid (N/64, M/128), 256 threads = 8 waves; wave w owns 16 rows x 64 cols
// (4 accumulators -> 4 WMMAs per A-fragment).
// ---------------------------------------------------------------------------
template<typename TA, typename TB, typename TC, bool TRANS_B>
__global__ __launch_bounds__(256) void wmma_gemm(
    const TA* __restrict__ A, int lda,
    const TB* __restrict__ Bp, int ldb,
    TC* __restrict__ C, int ldc, int K)
{
  __shared__ _Float16 As[128][40];
  __shared__ _Float16 Bs[64][40];    // [n_local][k_local]
  const int tid  = threadIdx.x;
  const int lane = tid & 31;
  const int wave = tid >> 5;
  const int m0   = blockIdx.y * 128;
  const int n0   = blockIdx.x * 64;
  const int lm   = lane & 15;
  const int lh   = lane >> 4;

  v8f acc[4] = {};

  for (int k0 = 0; k0 < K; k0 += 32) {
    #pragma unroll
    for (int i = 0; i < 16; ++i) {               // A tile 128x32
      int idx = i * 256 + tid;
      int r = idx >> 5, c = idx & 31;
      As[r][c] = conv_h(A[(size_t)(m0 + r) * lda + (k0 + c)]);
    }
    #pragma unroll
    for (int i = 0; i < 8; ++i) {                // B tile 64x32 -> Bs[n][k]
      int idx = i * 256 + tid;
      if (TRANS_B) {
        int r = idx >> 5, c = idx & 31;
        Bs[r][c] = conv_h(Bp[(size_t)(n0 + r) * ldb + (k0 + c)]);
      } else {
        int r = idx & 63, c = idx >> 6;          // r fastest -> coalesced along N
        Bs[r][c] = conv_h(Bp[(size_t)(k0 + c) * ldb + (n0 + r)]);
      }
    }
    __syncthreads();

    v16h a;
    #pragma unroll
    for (int e = 0; e < 16; ++e) {               // ISA A layout: k(e)=8*lh+e+(e>=8?8:0)
      int kk = 8 * lh + e + ((e & 8) ? 8 : 0);
      a[e] = As[wave * 16 + lm][kk];
    }
    #pragma unroll
    for (int ni = 0; ni < 4; ++ni) {
      v16h b;
      #pragma unroll
      for (int e = 0; e < 16; ++e) b[e] = Bs[ni * 16 + lm][8 * lh + e + ((e & 8) ? 8 : 0)];
      acc[ni] = __builtin_amdgcn_wmma_f32_16x16x32_f16(false, a, false, b, (short)0, acc[ni], false, false);
    }
    __syncthreads();
  }

  #pragma unroll
  for (int ni = 0; ni < 4; ++ni)
    #pragma unroll
    for (int v = 0; v < 8; ++v) {
      int row = m0 + wave * 16 + lh * 8 + v;
      C[(size_t)row * ldc + (n0 + ni * 16 + lm)] = (TC)acc[ni][v];
    }
}

// ---------------------------------------------------------------------------
// TDM helper: 2-D tensor_load_to_lds of a [tile_rows x 32] f16 tile whose rows
// are stride_elems apart in global memory, into a compact LDS tile (stride 32).
// Descriptor packing per CDNA5 ISA §8.3/8.4 (data_size=1 -> 2 bytes, type=2).
// 6-arg builtin form (clang-23 / therock-10.0): groups 2/3 + trailing group
// are zero for a 2-D tensor; last arg is cache policy.
// ---------------------------------------------------------------------------
__device__ __forceinline__ void tdm_load_tile_f16(
    uint32_t lds_off, const _Float16* gptr, int tile_rows, int stride_elems)
{
  uint64_t ga = (uint64_t)(uintptr_t)gptr;
  u32x4 g0;
  g0[0] = 1u;                                            // count=1 (user D#)
  g0[1] = lds_off;                                       // lds_addr (bytes)
  g0[2] = (uint32_t)ga;                                  // global_addr[31:0]
  g0[3] = (uint32_t)((ga >> 32) & 0x1FFFFFFu) | (2u << 30); // addr[56:32] | type=2
  i32x8 g1;
  g1[0] = 0x00010000;                                    // data_size=1 (2B), no mcast/pad
  g1[1] = (int)(((uint32_t)stride_elems & 0xFFFFu) << 16);          // tensor_dim0[15:0]
  g1[2] = (int)(((uint32_t)stride_elems >> 16) |
                (((uint32_t)tile_rows & 0xFFFFu) << 16));           // dim0[31:16] | dim1[15:0]
  g1[3] = (int)((((uint32_t)tile_rows >> 16) & 0xFFFFu) |
                (32u << 16));                                       // dim1[31:16] | tile_dim0=32
  g1[4] = (int)((uint32_t)tile_rows & 0xFFFFu);                     // tile_dim1 | tile_dim2=0
  g1[5] = (int)(uint32_t)stride_elems;                              // tensor_dim0_stride[31:0]
  g1[6] = 0;                                                        // stride0[47:32] | stride1[15:0]
  g1[7] = 0;
  i32x4 z4 = {};
  i32x8 z8 = {};
  __builtin_amdgcn_tensor_load_to_lds(g0, g1, z4, z4, z8, 0);
}

// ---------------------------------------------------------------------------
// Scores GEMM (all-f16 NT): C[M,N] = A[M,K] * W[N,K]^T, f32 out.
// Tiles streamed by the Tensor Data Mover into double-buffered LDS; wave 0
// issues the DMAs and pipelines them with s_wait_tensorcnt.
// Grid (N/64, M/128), 256 threads.
// ---------------------------------------------------------------------------
__global__ __launch_bounds__(256) void wmma_gemm_f16_tdm(
    const _Float16* __restrict__ A, int lda,
    const _Float16* __restrict__ W, int ldb,
    float* __restrict__ C, int ldc, int K)
{
  __shared__ _Float16 As[2][128][32];
  __shared__ _Float16 Bs[2][64][32];
  const int tid  = threadIdx.x;
  const int lane = tid & 31;
  const int wave = tid >> 5;
  const int m0   = blockIdx.y * 128;
  const int n0   = blockIdx.x * 64;
  const int lm   = lane & 15;
  const int lh   = lane >> 4;

  v8f acc[4] = {};
  const int nk = K >> 5;

  if (wave == 0) {   // prologue: stream first K-slab into buffer 0
    tdm_load_tile_f16((uint32_t)(uintptr_t)&As[0][0][0], A + (size_t)m0 * lda, 128, lda);
    tdm_load_tile_f16((uint32_t)(uintptr_t)&Bs[0][0][0], W + (size_t)n0 * ldb, 64, ldb);
  }

  for (int kt = 0; kt < nk; ++kt) {
    const int cur = kt & 1, nxt = cur ^ 1;
    if (wave == 0) {
      if (kt + 1 < nk) {
        const int k1 = (kt + 1) << 5;
        tdm_load_tile_f16((uint32_t)(uintptr_t)&As[nxt][0][0], A + (size_t)m0 * lda + k1, 128, lda);
        tdm_load_tile_f16((uint32_t)(uintptr_t)&Bs[nxt][0][0], W + (size_t)n0 * ldb + k1, 64, ldb);
        __builtin_amdgcn_s_wait_tensorcnt(2);   // current pair done (in-order), next in flight
      } else {
        __builtin_amdgcn_s_wait_tensorcnt(0);
      }
    }
    __syncthreads();

    v16h a;
    #pragma unroll
    for (int e = 0; e < 16; ++e) {
      int kk = 8 * lh + e + ((e & 8) ? 8 : 0);
      a[e] = As[cur][wave * 16 + lm][kk];
    }
    #pragma unroll
    for (int ni = 0; ni < 4; ++ni) {
      v16h b;
      #pragma unroll
      for (int e = 0; e < 16; ++e) b[e] = Bs[cur][ni * 16 + lm][8 * lh + e + ((e & 8) ? 8 : 0)];
      acc[ni] = __builtin_amdgcn_wmma_f32_16x16x32_f16(false, a, false, b, (short)0, acc[ni], false, false);
    }
    __syncthreads();
  }

  #pragma unroll
  for (int ni = 0; ni < 4; ++ni)
    #pragma unroll
    for (int v = 0; v < 8; ++v) {
      int row = m0 + wave * 16 + lh * 8 + v;
      C[(size_t)row * ldc + (n0 + ni * 16 + lm)] = acc[ni][v];
    }
}

// ---------------------------------------------------------------------------
template<typename TOut>
__global__ __launch_bounds__(256) void rmsnorm_k(
    const float* __restrict__ src, int sstride,
    TOut* __restrict__ dst, int dstride,
    int n, const float* __restrict__ w)
{
  __shared__ float red[256];
  const int row = blockIdx.x;
  const float* s = src + (size_t)row * sstride;
  TOut* d = dst + (size_t)row * dstride;
  float acc = 0.f;
  for (int i = threadIdx.x; i < n; i += 256) { float v = s[i]; acc += v * v; }
  red[threadIdx.x] = acc; __syncthreads();
  for (int off = 128; off > 0; off >>= 1) {
    if (threadIdx.x < off) red[threadIdx.x] += red[threadIdx.x + off];
    __syncthreads();
  }
  const float scale = rsqrtf(red[0] / (float)n + 1e-6f);
  for (int i = threadIdx.x; i < n; i += 256) d[i] = (TOut)(s[i] * scale * w[i]);
}

__global__ __launch_bounds__(256) void rope_k(
    const float* __restrict__ src, int sstride, int shead,
    _Float16* __restrict__ dst, int dstride, long dhead,
    const float* __restrict__ cosb, const float* __restrict__ sinb, int nheads)
{
  const int idx = blockIdx.x * blockDim.x + threadIdx.x;
  const int i   = idx & 31;
  const int t   = idx >> 5;
  const int h   = t % nheads;
  const int row = t / nheads;
  const int pos = row & (S_ - 1);     // row = b*S + s
  const float c  = cosb[pos * 32 + i];
  const float sn = sinb[pos * 32 + i];
  const float* p = src + (size_t)row * sstride + (size_t)h * shead;
  const float x0 = p[2 * i], x1 = p[2 * i + 1];
  _Float16* q = dst + (size_t)h * dhead + (size_t)row * dstride;
  q[2 * i]     = (_Float16)(x0 * c - x1 * sn);
  q[2 * i + 1] = (_Float16)(x0 * sn + x1 * c);
}

__global__ __launch_bounds__(256) void softmax_k(
    float* __restrict__ sc, const float* __restrict__ mask, float scale)
{
  __shared__ float red[256];
  const int row = blockIdx.x;
  float* r = sc + (size_t)row * S_;
  const float* m = mask + (size_t)row * S_;
  float lmax = -1e30f;
  for (int t = threadIdx.x; t < S_; t += 256) {
    float v = r[t] * scale + m[t];
    r[t] = v;
    lmax = fmaxf(lmax, v);
  }
  red[threadIdx.x] = lmax; __syncthreads();
  for (int off = 128; off > 0; off >>= 1) {
    if (threadIdx.x < off) red[threadIdx.x] = fmaxf(red[threadIdx.x], red[threadIdx.x + off]);
    __syncthreads();
  }
  const float gmax = red[0]; __syncthreads();
  float lsum = 0.f;
  for (int t = threadIdx.x; t < S_; t += 256) {
    float e = __expf(r[t] - gmax);
    r[t] = e; lsum += e;
  }
  red[threadIdx.x] = lsum; __syncthreads();
  for (int off = 128; off > 0; off >>= 1) {
    if (threadIdx.x < off) red[threadIdx.x] += red[threadIdx.x + off];
    __syncthreads();
  }
  const float inv = 1.f / red[0];
  for (int t = threadIdx.x; t < S_; t += 256) r[t] *= inv;
}

extern "C" void kernel_launch(void* const* d_in, const int* in_sizes, int n_in,
                              void* d_out, int out_size, void* d_ws, size_t ws_size,
                              hipStream_t stream)
{
  (void)in_sizes; (void)n_in; (void)out_size; (void)ws_size;
  const float* x     = (const float*)d_in[0];
  const float* cosb  = (const float*)d_in[1];
  const float* sinb  = (const float*)d_in[2];
  const float* mask  = (const float*)d_in[3];
  const float* wq_a  = (const float*)d_in[4];
  const float* qln   = (const float*)d_in[5];
  const float* wq_b  = (const float*)d_in[6];
  const float* wkv_a = (const float*)d_in[7];
  const float* kvln  = (const float*)d_in[8];
  const float* wkv_b = (const float*)d_in[9];
  const float* wo    = (const float*)d_in[10];
  float* out = (float*)d_out;

  const int M = B_ * S_;  // 4096 rows of (b,s)
  char* p = (char*)d_ws;
  float*    q_lat = (float*)p;     p += (size_t)M * QLORA * 4;          // 25.2 MB
  float*    q_f32 = (float*)p;     p += (size_t)M * (H_ * QKH) * 4;     // 50.3 MB
  float*    kv    = (float*)p;     p += (size_t)M * 576 * 4;            //  9.4 MB
  _Float16* q_big = (_Float16*)p;  p += (size_t)H_ * M * 576 * 2;       // 75.5 MB  [h][row][576]
  _Float16* k_big = (_Float16*)p;  p += (size_t)M * 576 * 2;            //  4.7 MB  [row][576]
  _Float16* ctx   = (_Float16*)p;  p += (size_t)M * H_ * 512 * 2;       // 67.1 MB  [row][h*512]
  // q_f32 dead after step 8 -> reuse for scores (16 MB) + out_heads (33.5 MB)
  float* scores    = q_f32;
  float* out_heads = q_f32 + (size_t)S_ * S_;

  dim3 blk(256);
  const float SCALE = 0.07216878364870323f;  // 192^-0.5

  // 1) q_lat = x @ wq_a^T            [4096,1536]
  wmma_gemm<float, float, float, true><<<dim3(QLORA/64, M/128), blk, 0, stream>>>(
      x, DIM_, wq_a, DIM_, q_lat, QLORA, DIM_);
  // 2) RMSNorm(q_lat) in place
  rmsnorm_k<float><<<M, blk, 0, stream>>>(q_lat, QLORA, q_lat, QLORA, QLORA, qln);
  // 3) q = q_lat @ wq_b^T            [4096,3072]
  wmma_gemm<float, float, float, true><<<dim3(3072/64, M/128), blk, 0, stream>>>(
      q_lat, QLORA, wq_b, QLORA, q_f32, H_*QKH, QLORA);
  // 4) kv = x @ wkv_a^T              [4096,576]
  wmma_gemm<float, float, float, true><<<dim3(576/64, M/128), blk, 0, stream>>>(
      x, DIM_, wkv_a, DIM_, kv, 576, DIM_);
  // 5) kv_cache = RMSNorm(kv[:, :512]) -> f16 k_big cols 0..511
  rmsnorm_k<_Float16><<<M, blk, 0, stream>>>(kv, 576, k_big, 576, KVLORA, kvln);
  // 6) RoPE(k_pe) -> f16 k_big cols 512..575
  rope_k<<<(M * 32) / 256, blk, 0, stream>>>(
      kv + KVLORA, 576, 0, k_big + KVLORA, 576, 0L, cosb, sinb, 1);
  // 7) RoPE(q_pe) -> f16 q_big[h] cols 512..575
  rope_k<<<(M * H_ * 32) / 256, blk, 0, stream>>>(
      q_f32 + NOPE_, H_*QKH, QKH, q_big + KVLORA, 576, (long)M * 576, cosb, sinb, H_);
  // 8) absorbed q: q_nope_h @ wkv_b3[h,:128,:] -> f16 q_big[h] cols 0..511
  for (int h = 0; h < H_; ++h) {
    wmma_gemm<float, float, _Float16, false><<<dim3(KVLORA/64, M/128), blk, 0, stream>>>(
        q_f32 + h * QKH, H_*QKH,
        wkv_b + (size_t)h * 256 * 512, 512,
        q_big + (size_t)h * M * 576, 576, NOPE_);
  }
  // 9) attention per (b,h): scores (TDM-fed WMMA) -> softmax -> ctx
  for (int b = 0; b < B_; ++b) {
    const _Float16* Kb = k_big + (size_t)b * S_ * 576;
    for (int h = 0; h < H_; ++h) {
      const _Float16* Qh = q_big + (size_t)h * M * 576 + (size_t)b * S_ * 576;
      wmma_gemm_f16_tdm<<<dim3(S_/64, S_/128), blk, 0, stream>>>(
          Qh, 576, Kb, 576, scores, S_, 576);
      softmax_k<<<S_, blk, 0, stream>>>(scores, mask, SCALE);
      wmma_gemm<float, _Float16, _Float16, false><<<dim3(KVLORA/64, S_/128), blk, 0, stream>>>(
          scores, S_, Kb, 576,
          ctx + (size_t)b * S_ * H_ * 512 + (size_t)h * 512, H_*512, S_);
    }
  }
  // 10) per-head output: ctx_h @ wv_h^T -> out_heads[:, h*128 : h*128+128]
  for (int h = 0; h < H_; ++h) {
    wmma_gemm<_Float16, float, float, true><<<dim3(128/64, M/128), blk, 0, stream>>>(
        ctx + (size_t)h * 512, H_*512,
        wkv_b + (size_t)(h * 256 + NOPE_) * 512, 512,
        out_heads + (size_t)h * 128, DIM_, KVLORA);
  }
  // 11) final projection: out = out_heads @ wo^T   [4096,2048]
  wmma_gemm<float, float, float, true><<<dim3(DIM_/64, M/128), blk, 0, stream>>>(
      out_heads, DIM_, wo, DIM_, out, DIM_, DIM_);
}
